// GNN_15350213116754
// MI455X (gfx1250) — matrix operand order repaired and checked by reading, not compiled, not run on hardware
//
#include <hip/hip_runtime.h>
#include <hip/hip_bf16.h>
#include <math.h>

#define Hdim 128
#define H2   256

typedef __attribute__((ext_vector_type(16))) __bf16 v16bf;
typedef __attribute__((ext_vector_type(8)))  float  v8f;

__device__ __forceinline__ float gelu_exact(float x) {
  // torch GELU default: exact erf form
  return 0.5f * x * (1.0f + erff(x * 0.70710678118654752440f));
}

// ---------------- elementwise / graph kernels ----------------

__global__ void k_fill(float* __restrict__ p, float v, long n) {
  long i = blockIdx.x * (long)blockDim.x + threadIdx.x;
  if (i < n) p[i] = v;
}

__global__ void k_deg(float* __restrict__ deg, const int* __restrict__ dst, int E) {
  int i = blockIdx.x * blockDim.x + threadIdx.x;
  if (i < E) atomicAdd(&deg[dst[i]], 1.0f);
}

__global__ void k_rsqrt(float* __restrict__ p, int n) {
  int i = blockIdx.x * blockDim.x + threadIdx.x;
  if (i < n) p[i] = rsqrtf(fmaxf(p[i], 1.0f));
}

__global__ void k_counts(float* __restrict__ counts, const int* __restrict__ batch, int n) {
  int i = blockIdx.x * blockDim.x + threadIdx.x;
  if (i < n) atomicAdd(&counts[batch[i]], 1.0f);
}

__global__ void k_invc(float* __restrict__ invc, const float* __restrict__ counts, int g) {
  int i = blockIdx.x * blockDim.x + threadIdx.x;
  if (i < g) invc[i] = 1.0f / fmaxf(counts[i], 1.0f);
}

__global__ void k_vninit(float* __restrict__ vn, const float* __restrict__ emb, long n) {
  long i = blockIdx.x * (long)blockDim.x + threadIdx.x;
  if (i < n) vn[i] = emb[i & (Hdim - 1)];
}

__global__ void k_addvn(float* __restrict__ x, const float* __restrict__ vn,
                        const int* __restrict__ batch, long n) {
  long i = blockIdx.x * (long)blockDim.x + threadIdx.x;
  if (i >= n) return;
  int node = (int)(i >> 7);
  int c    = (int)(i & 127);
  x[i] += vn[(size_t)batch[node] * Hdim + c];
}

__global__ void k_add(float* __restrict__ dst, const float* __restrict__ a, long n) {
  long i = blockIdx.x * (long)blockDim.x + threadIdx.x;
  if (i < n) dst[i] += a[i];
}

__global__ void k_edge_scatter(float* __restrict__ agg, const float* __restrict__ h,
                               const int* __restrict__ src, const int* __restrict__ dst,
                               const float* __restrict__ dinv, long total) {
  long gid = blockIdx.x * (long)blockDim.x + threadIdx.x;
  if (gid >= total) return;
  int e = (int)(gid >> 5);   // 32 quads of 4 columns per edge (H=128)
  int q = (int)(gid & 31);
  int s = src[e], d = dst[e];
  float nrm = dinv[s] * dinv[d];
  float4 hv = *(const float4*)(h + (size_t)s * Hdim + q * 4);
  float* ap = agg + (size_t)d * Hdim + q * 4;
  atomicAdd(ap + 0, hv.x * nrm);
  atomicAdd(ap + 1, hv.y * nrm);
  atomicAdd(ap + 2, hv.z * nrm);
  atomicAdd(ap + 3, hv.w * nrm);
}

// x = agg + h*dinv^2 (self loop) + conv_b + x (residual)
__global__ void k_combine(float* __restrict__ x, const float* __restrict__ agg,
                          const float* __restrict__ h, const float* __restrict__ dinv,
                          const float* __restrict__ cb, long n) {
  long i = blockIdx.x * (long)blockDim.x + threadIdx.x;
  if (i >= n) return;
  int node = (int)(i >> 7);
  int c    = (int)(i & 127);
  float di = dinv[node];
  x[i] = agg[i] + h[i] * di * di + cb[c] + x[i];
}

__global__ void k_pool(float* __restrict__ pooled, const float* __restrict__ x,
                       const int* __restrict__ batch, long total) {
  long gid = blockIdx.x * (long)blockDim.x + threadIdx.x;
  if (gid >= total) return;
  int node = (int)(gid >> 5);
  int q    = (int)(gid & 31);
  int g = batch[node];
  float4 v = *(const float4*)(x + (size_t)node * Hdim + q * 4);
  float* pp = pooled + (size_t)g * Hdim + q * 4;
  atomicAdd(pp + 0, v.x);
  atomicAdd(pp + 1, v.y);
  atomicAdd(pp + 2, v.z);
  atomicAdd(pp + 3, v.w);
}

__global__ void k_vnplus(float* __restrict__ vn, const float* __restrict__ pooled,
                         const float* __restrict__ invc, long n) {
  long i = blockIdx.x * (long)blockDim.x + threadIdx.x;
  if (i < n) vn[i] += pooled[i] * invc[i >> 7];
}

// ---------------- BatchNorm (training-mode batch stats, biased var) ----------------

// blockDim.x == C; each block accumulates a row-chunk, then atomics into stats[0:C]=sum, stats[C:2C]=sumsq
__global__ void k_colstats(const float* __restrict__ x, float* __restrict__ stats,
                           int rows, int rowsPer) {
  int C = blockDim.x;
  int c = threadIdx.x;
  long r0 = (long)blockIdx.x * rowsPer;
  long r1 = r0 + rowsPer; if (r1 > rows) r1 = rows;
  float s = 0.0f, s2 = 0.0f;
  for (long r = r0; r < r1; ++r) {
    float v = x[r * C + c];
    s += v; s2 += v * v;
  }
  atomicAdd(&stats[c], s);
  atomicAdd(&stats[C + c], s2);
}

__global__ void k_bn(float* __restrict__ x, const float* __restrict__ stats,
                     const float* __restrict__ g, const float* __restrict__ b,
                     int rows, int C, int fuseGelu, long n) {
  long i = blockIdx.x * (long)blockDim.x + threadIdx.x;
  if (i >= n) return;
  int c = (int)(i % C);
  float invR = 1.0f / (float)rows;
  float m  = stats[c] * invR;
  float vv = stats[C + c] * invR - m * m;
  float y = (x[i] - m) * rsqrtf(vv + 1e-5f) * g[c] + b[c];
  if (fuseGelu) y = gelu_exact(y);
  x[i] = y;
}

// ---------------- weight transpose+convert: W[K][Nout] f32 -> Wt[Nout][K] bf16 ----------------

__global__ void k_wt(__bf16* __restrict__ Wt, const float* __restrict__ W, int K, int Nout) {
  long i = blockIdx.x * (long)blockDim.x + threadIdx.x;
  long tot = (long)K * Nout;
  if (i >= tot) return;
  int n = (int)(i % Nout);
  int k = (int)(i / Nout);
  Wt[(size_t)n * K + k] = (__bf16)W[i];
}

// ---------------- WMMA GEMM: out[M][Nout] = A[M][K] @ W + bias (optional GELU) ----------------
// 1x4 register blocking: each wave computes a 16x64 output strip (4 accumulators),
// reusing the f32->bf16 converted A fragment across 4 WMMAs per K-step.
// block = 256 threads = 8 waves:
//   wave w -> row tile  tm = blockIdx.x*4 + (w>>1)   (clamped; stores guarded, wave-uniform)
//             col tiles tn0 = blockIdx.y*8 + (w&1)*4 .. +3
// grid.x = ceil(M16/4), grid.y = Nout/128.
__device__ __forceinline__ void store_tile(float* __restrict__ out, int Nout, int rowBase,
                                           int col, v8f acc, float bb, int fuseGelu) {
  float* op = out + (size_t)rowBase * Nout + col;
#pragma unroll
  for (int r = 0; r < 8; ++r) {
    float v = acc[r] + bb;
    if (fuseGelu) v = gelu_exact(v);
    op[(size_t)r * Nout] = v;
  }
}

__global__ void __launch_bounds__(256) k_gemm(const float* __restrict__ A,
                                              const __bf16* __restrict__ Wt,
                                              const float* __restrict__ bias,
                                              float* __restrict__ out,
                                              int M16, int K, int Nout, int fuseGelu) {
  const int lane = threadIdx.x & 31;
  const int wave = threadIdx.x >> 5;
  const int half = lane >> 4;   // K-half select
  const int l16  = lane & 15;   // A row / B column within tile

  const int tmg   = blockIdx.x * 4 + (wave >> 1);   // wave-uniform
  const bool valid = (tmg < M16);
  const int tm    = valid ? tmg : (M16 - 1);        // clamp: loads stay in-bounds
  const int tn0   = blockIdx.y * 8 + (wave & 1) * 4;

  const float*  ap  = A + (size_t)(tm * 16 + l16) * K;
  const __bf16* bp0 = Wt + (size_t)((tn0 * 16) + l16) * K + half * 16;
  const __bf16* bp1 = bp0 + (size_t)16 * K;
  const __bf16* bp2 = bp0 + (size_t)32 * K;
  const __bf16* bp3 = bp0 + (size_t)48 * K;

  v8f acc0 = {}, acc1 = {}, acc2 = {}, acc3 = {};
  for (int ks = 0; ks < K; ks += 32) {
    // A fragment: lane holds K = half*16 + {0..15}; convert f32->bf16 in-register
    const float* alo = ap + ks + half * 16;
    const float* ahi = alo + 8;
    v16bf a;
#pragma unroll
    for (int j = 0; j < 8; ++j) {
      a[j]     = (__bf16)alo[j];
      a[8 + j] = (__bf16)ahi[j];
    }
    // B fragments: lane holds K = half*16 + {0..15}, contiguous 32B loads
    v16bf b0 = *(const v16bf*)(bp0 + ks);
    v16bf b1 = *(const v16bf*)(bp1 + ks);
    v16bf b2 = *(const v16bf*)(bp2 + ks);
    v16bf b3 = *(const v16bf*)(bp3 + ks);
    acc0 = __builtin_amdgcn_wmma_f32_16x16x32_bf16(false, a, false, b0, (short)0, acc0, false, false);
    acc1 = __builtin_amdgcn_wmma_f32_16x16x32_bf16(false, a, false, b1, (short)0, acc1, false, false);
    acc2 = __builtin_amdgcn_wmma_f32_16x16x32_bf16(false, a, false, b2, (short)0, acc2, false, false);
    acc3 = __builtin_amdgcn_wmma_f32_16x16x32_bf16(false, a, false, b3, (short)0, acc3, false, false);
  }

  if (valid) {  // wave-uniform guard; EXEC was all-ones through all WMMAs
    const int rowBase = tm * 16 + half * 8;
    const int c0 = tn0 * 16 + l16;
    float bb0 = bias ? bias[c0]      : 0.0f;
    float bb1 = bias ? bias[c0 + 16] : 0.0f;
    float bb2 = bias ? bias[c0 + 32] : 0.0f;
    float bb3 = bias ? bias[c0 + 48] : 0.0f;
    store_tile(out, Nout, rowBase, c0,      acc0, bb0, fuseGelu);
    store_tile(out, Nout, rowBase, c0 + 16, acc1, bb1, fuseGelu);
    store_tile(out, Nout, rowBase, c0 + 32, acc2, bb2, fuseGelu);
    store_tile(out, Nout, rowBase, c0 + 48, acc3, bb3, fuseGelu);
  }
}

// ---------------- host orchestration ----------------

static inline int cdiv(long a, long b) { return (int)((a + b - 1) / b); }

extern "C" void kernel_launch(void* const* d_in, const int* in_sizes, int n_in,
                              void* d_out, int out_size, void* d_ws, size_t ws_size,
                              hipStream_t stream) {
  const float* x_in    = (const float*)d_in[0];
  const int*   eidx    = (const int*)d_in[1];
  const int*   batch   = (const int*)d_in[2];
  const float* pre_W1  = (const float*)d_in[3];
  const float* pre_b1  = (const float*)d_in[4];
  const float* pre_W2  = (const float*)d_in[5];
  const float* pre_b2  = (const float*)d_in[6];
  const float* conv_W  = (const float*)d_in[7];
  const float* conv_b  = (const float*)d_in[8];
  const float* bn_g    = (const float*)d_in[9];
  const float* bn_b    = (const float*)d_in[10];
  const float* ffn_W1  = (const float*)d_in[11];
  const float* ffn_b1  = (const float*)d_in[12];
  const float* ffn_W2  = (const float*)d_in[13];
  const float* ffn_b2  = (const float*)d_in[14];
  const float* vn_W1   = (const float*)d_in[15];
  const float* vn_b1   = (const float*)d_in[16];
  const float* vn_g1   = (const float*)d_in[17];
  const float* vn_be1  = (const float*)d_in[18];
  const float* vn_W2   = (const float*)d_in[19];
  const float* vn_b2   = (const float*)d_in[20];
  const float* vn_g2   = (const float*)d_in[21];
  const float* vn_be2  = (const float*)d_in[22];
  const float* vn_emb  = (const float*)d_in[23];
  const float* post_W1 = (const float*)d_in[24];
  const float* post_b1 = (const float*)d_in[25];
  const float* post_W2 = (const float*)d_in[26];
  const float* post_b2 = (const float*)d_in[27];

  const int N    = in_sizes[0] / Hdim;
  const int E    = in_sizes[1] / 2;
  const int G    = out_size / Hdim;
  const int HOPS = in_sizes[7] / (Hdim * Hdim);
  const int* src = eidx;
  const int* dst = eidx + E;

  // workspace carve
  float* B      = (float*)d_ws;                    // N*H   : running x
  float* Cb     = B + (size_t)N * Hdim;            // N*H   : GEMM out / FFN out
  float* A2     = Cb + (size_t)N * Hdim;           // N*2H  : FFN hidden; first N*H doubles as agg
  float* dinv   = A2 + (size_t)N * H2;             // N
  float* counts = dinv + N;                        // G
  float* invc   = counts + G;                      // G
  float* vnb    = invc + G;                        // G*H
  float* t1     = vnb + (size_t)G * Hdim;          // G*2H
  float* t2     = t1 + (size_t)G * H2;             // G*H
  float* pooled = t2 + (size_t)G * Hdim;           // G*H
  float* stats  = pooled + (size_t)G * Hdim;       // 512
  __bf16* Wt    = (__bf16*)(stats + 512);          // 256*256 bf16

  const long nNH = (long)N * Hdim;
  const long escat = (long)E * (Hdim / 4);
  const long pscat = (long)N * (Hdim / 4);

  // ---- graph prep: degrees (self loop counts as 1), inv-sqrt, graph counts ----
  k_fill<<<cdiv(N, 256), 256, 0, stream>>>(dinv, 1.0f, N);
  k_deg<<<cdiv(E, 256), 256, 0, stream>>>(dinv, dst, E);
  k_rsqrt<<<cdiv(N, 256), 256, 0, stream>>>(dinv, N);
  k_fill<<<1, 256, 0, stream>>>(counts, 0.0f, G);
  k_counts<<<cdiv(N, 256), 256, 0, stream>>>(counts, batch, N);
  k_invc<<<1, 256, 0, stream>>>(invc, counts, G);

  const int M16  = N / 16;                    // 6250 row tiles (N multiple of 16)
  const int GM16 = G / 16;                    // 4 row tiles for small GEMMs
  dim3 g1(cdiv(M16, 4), 1), g2(cdiv(M16, 4), 2);   // Nout=128 / Nout=256
  dim3 gs1(cdiv(GM16, 4), 1), gs2(cdiv(GM16, 4), 2);

  // ---- pre-FFNN: x = gelu(gelu(x@W1+b1)@W2+b2) ----
  k_wt<<<cdiv((long)Hdim * H2, 256), 256, 0, stream>>>(Wt, pre_W1, Hdim, H2);
  k_gemm<<<g2, 256, 0, stream>>>(x_in, Wt, pre_b1, A2, M16, Hdim, H2, 1);
  k_wt<<<cdiv((long)H2 * Hdim, 256), 256, 0, stream>>>(Wt, pre_W2, H2, Hdim);
  k_gemm<<<g1, 256, 0, stream>>>(A2, Wt, pre_b2, B, M16, H2, Hdim, 1);

  // ---- virtual node init ----
  k_vninit<<<cdiv((long)G * Hdim, 256), 256, 0, stream>>>(vnb, vn_emb, (long)G * Hdim);
  float* vn = vnb;

  for (int i = 0; i < HOPS; ++i) {
    // x = x + vn[batch]
    k_addvn<<<cdiv(nNH, 256), 256, 0, stream>>>(B, vn, batch, nNH);

    // GCN: h = x @ conv_W[i]
    k_wt<<<cdiv((long)Hdim * Hdim, 256), 256, 0, stream>>>(Wt, conv_W + (size_t)i * Hdim * Hdim, Hdim, Hdim);
    k_gemm<<<g1, 256, 0, stream>>>(B, Wt, nullptr, Cb, M16, Hdim, Hdim, 0);
    // agg = scatter(h[src]*norm -> dst)
    k_fill<<<cdiv(nNH, 256), 256, 0, stream>>>(A2, 0.0f, nNH);
    k_edge_scatter<<<cdiv(escat, 256), 256, 0, stream>>>(A2, Cb, src, dst, dinv, escat);
    // x = agg + h*dinv^2 + conv_b + x
    k_combine<<<cdiv(nNH, 256), 256, 0, stream>>>(B, A2, Cb, dinv, conv_b + (size_t)i * Hdim, nNH);

    // BatchNorm over N rows
    k_fill<<<1, 512, 0, stream>>>(stats, 0.0f, 512);
    k_colstats<<<cdiv(N, 256), Hdim, 0, stream>>>(B, stats, N, 256);
    k_bn<<<cdiv(nNH, 256), 256, 0, stream>>>(B, stats, bn_g + (size_t)i * Hdim, bn_b + (size_t)i * Hdim,
                                             N, Hdim, 0, nNH);

    // FFN: x = gelu(gelu(x@W1+b1)@W2+b2) + x
    k_wt<<<cdiv((long)Hdim * H2, 256), 256, 0, stream>>>(Wt, ffn_W1 + (size_t)i * Hdim * H2, Hdim, H2);
    k_gemm<<<g2, 256, 0, stream>>>(B, Wt, ffn_b1 + (size_t)i * H2, A2, M16, Hdim, H2, 1);
    k_wt<<<cdiv((long)H2 * Hdim, 256), 256, 0, stream>>>(Wt, ffn_W2 + (size_t)i * H2 * Hdim, H2, Hdim);
    k_gemm<<<g1, 256, 0, stream>>>(A2, Wt, ffn_b2 + (size_t)i * Hdim, Cb, M16, H2, Hdim, 1);
    k_add<<<cdiv(nNH, 256), 256, 0, stream>>>(B, Cb, nNH);

    if (i < HOPS - 1) {
      // vn = vn + mean-pool(x); vn = vn_mlp(vn)
      k_fill<<<cdiv((long)G * Hdim, 256), 256, 0, stream>>>(pooled, 0.0f, (long)G * Hdim);
      k_pool<<<cdiv(pscat, 256), 256, 0, stream>>>(pooled, B, batch, pscat);
      k_vnplus<<<cdiv((long)G * Hdim, 256), 256, 0, stream>>>(vn, pooled, invc, (long)G * Hdim);

      k_wt<<<cdiv((long)Hdim * H2, 256), 256, 0, stream>>>(Wt, vn_W1 + (size_t)i * Hdim * H2, Hdim, H2);
      k_gemm<<<gs2, 256, 0, stream>>>(vn, Wt, vn_b1 + (size_t)i * H2, t1, GM16, Hdim, H2, 0);
      k_fill<<<1, 512, 0, stream>>>(stats, 0.0f, 512);
      k_colstats<<<1, H2, 0, stream>>>(t1, stats, G, G);
      k_bn<<<cdiv((long)G * H2, 256), 256, 0, stream>>>(t1, stats, vn_g1 + (size_t)i * H2,
                                                        vn_be1 + (size_t)i * H2, G, H2, 1, (long)G * H2);

      k_wt<<<cdiv((long)H2 * Hdim, 256), 256, 0, stream>>>(Wt, vn_W2 + (size_t)i * H2 * Hdim, H2, Hdim);
      k_gemm<<<gs1, 256, 0, stream>>>(t1, Wt, vn_b2 + (size_t)i * Hdim, t2, GM16, H2, Hdim, 0);
      k_fill<<<1, 512, 0, stream>>>(stats, 0.0f, 512);
      k_colstats<<<1, Hdim, 0, stream>>>(t2, stats, G, G);
      k_bn<<<cdiv((long)G * Hdim, 256), 256, 0, stream>>>(t2, stats, vn_g2 + (size_t)i * Hdim,
                                                          vn_be2 + (size_t)i * Hdim, G, Hdim, 1, (long)G * Hdim);
      vn = t2;
    }
  }

  // ---- final: global add pool + post-FFNN -> d_out (G x H f32) ----
  k_fill<<<cdiv((long)G * Hdim, 256), 256, 0, stream>>>(pooled, 0.0f, (long)G * Hdim);
  k_pool<<<cdiv(pscat, 256), 256, 0, stream>>>(pooled, B, batch, pscat);
  k_wt<<<cdiv((long)Hdim * H2, 256), 256, 0, stream>>>(Wt, post_W1, Hdim, H2);
  k_gemm<<<gs2, 256, 0, stream>>>(pooled, Wt, post_b1, t1, GM16, Hdim, H2, 1);
  k_wt<<<cdiv((long)H2 * Hdim, 256), 256, 0, stream>>>(Wt, post_W2, H2, Hdim);
  k_gemm<<<gs1, 256, 0, stream>>>(t1, Wt, post_b2, (float*)d_out, GM16, H2, Hdim, 1);
}
